// BidirectionalAntiAttention_15625091023578
// MI455X (gfx1250) — matrix operand order repaired and checked by reading, not compiled, not run on hardware
//
#include <hip/hip_runtime.h>
#include <hip/hip_bf16.h>
#include <cstdint>
#include <cstddef>

// ---------------- problem dims (compile-time) ----------------
#define B_DIM 4
#define L_DIM 2048
#define D_DIM 768
#define R_DIM 32
#define P_DIM 496           // R*(R-1)/2
#define KPAD  512           // P padded to WMMA K granularity
#define MTOT  (B_DIM * L_DIM)   // 8192 rows
#define E_DIM (3 * D_DIM)       // 2304 gate K

typedef __attribute__((ext_vector_type(16))) _Float16 v16h;
typedef __attribute__((ext_vector_type(8)))  _Float16 v8h;
typedef __attribute__((ext_vector_type(8)))  float    v8f;

union AFrag { v16h v; v8h h[2]; };

// ---------------- compile-time triu index table ----------------
struct IUTab { unsigned char a[P_DIM]; unsigned char b[P_DIM]; };
constexpr IUTab make_iu() {
  IUTab t{};
  int k = 0;
  for (int i = 0; i < R_DIM; ++i)
    for (int j = i + 1; j < R_DIM; ++j) { t.a[k] = (unsigned char)i; t.b[k] = (unsigned char)j; ++k; }
  return t;
}
__constant__ IUTab c_iu = make_iu();

// ---------------- K0: transpose weight [K][N] f32 -> [N][Kdst] f16 (zero-pad K) ----
__global__ __launch_bounds__(256)
void transpose_w_kernel(const float* __restrict__ W, _Float16* __restrict__ BT,
                        int Ksrc, int Kdst, int N) {
  int idx = blockIdx.x * 256 + threadIdx.x;
  if (idx >= N * Kdst) return;
  int n = idx / Kdst, k = idx % Kdst;
  BT[(size_t)n * Kdst + k] = (k < Ksrc) ? (_Float16)W[(size_t)k * N + n] : (_Float16)0.f;
}

// ---------------- K1: z = x @ W_dr + b_dr  (8 rows / block) ----------------
__global__ __launch_bounds__(256)
void zproj_kernel(const float* __restrict__ x, const float* __restrict__ Wdr,
                  const float* __restrict__ bdr, float* __restrict__ z) {
  __shared__ float xs[8][D_DIM];
  const int rowBase = blockIdx.x * 8;
  for (int i = threadIdx.x; i < 8 * D_DIM; i += 256) {
    int rr = i / D_DIM, dd = i % D_DIM;
    xs[rr][dd] = x[(size_t)(rowBase + rr) * D_DIM + dd];
  }
  __syncthreads();
  const int rr = threadIdx.x >> 5;   // row in block
  const int r  = threadIdx.x & 31;   // output channel
  float acc = bdr[r];
  #pragma unroll 4
  for (int d = 0; d < D_DIM; ++d)
    acc = fmaf(xs[rr][d], Wdr[d * R_DIM + r], acc);
  z[(size_t)(rowBase + rr) * R_DIM + r] = acc;
}

// ---------------- K2: summed normalized pluecker vectors per (dir,row) -----------
// rcnt[m] stores the *reciprocal* count (0 when no valid delta), so the GEMM
// epilogue needs no division at all.
__global__ __launch_bounds__(256)
void plucker_kernel(const float* __restrict__ z, _Float16* __restrict__ S,
                    float* __restrict__ rcnt, int dir) {
  __shared__ float zl[R_DIM], zr[R_DIM];
  __shared__ float pt[P_DIM];
  __shared__ float sacc[KPAD];
  __shared__ float red[256];
  const int m = blockIdx.x;           // 0..MTOT-1
  const int l = m & (L_DIM - 1);
  const int t = threadIdx.x;
  if (t < R_DIM) zl[t] = z[(size_t)m * R_DIM + t];
  for (int j = t; j < KPAD; j += 256) sacc[j] = 0.f;
  __syncthreads();
  int count = 0;
  #pragma unroll
  for (int di = 0; di < 4; ++di) {
    const int delta = 1 << di;        // 1,2,4,8
    const int partner = (dir == 0) ? (l + delta) : (l - delta);
    if (partner < 0 || partner >= L_DIM) continue;   // uniform across block
    ++count;
    __syncthreads();                  // protect zr/pt/red from previous iteration
    if (t < R_DIM) zr[t] = z[(size_t)(m - l + partner) * R_DIM + t];
    __syncthreads();
    float lss = 0.f;
    for (int j = t; j < P_DIM; j += 256) {
      const int ia = c_iu.a[j], ib = c_iu.b[j];
      const float p = zl[ia] * zr[ib] - zl[ib] * zr[ia];
      pt[j] = p;
      lss += p * p;
    }
    red[t] = lss; __syncthreads();
    for (int s = 128; s > 0; s >>= 1) { if (t < s) red[t] += red[t + s]; __syncthreads(); }
    const float inv = 1.0f / fmaxf(sqrtf(red[0]), 1e-8f);
    for (int j = t; j < P_DIM; j += 256) sacc[j] += pt[j] * inv;   // same-thread pt reuse
  }
  __syncthreads();
  for (int j = t; j < KPAD; j += 256)
    S[(size_t)m * KPAD + j] = (_Float16)sacc[j];
  if (t == 0) rcnt[m] = (count > 0) ? (1.0f / (float)count) : 0.0f;
}

// ---------------- K3: templated WMMA GEMM, C[M][ldc] (+)= A[M][KTOT] * BT[N][ldb]^T
// MODE 0: C = acc + bias   MODE 1: C += acc   MODE 2: C = acc*rcnt + (rcnt?bias:0)
template <int A_F16, int MODE, int KTOT>
__global__ __launch_bounds__(128)
void gemm_wmma_kernel(const void* __restrict__ Av, int lda,
                      const _Float16* __restrict__ BT, int ldb,
                      const float* __restrict__ bias, const float* __restrict__ rcnt,
                      float* __restrict__ C, int ldc) {
  const int lane  = threadIdx.x & 31;
  const int wave  = threadIdx.x >> 5;
  const int mBase = blockIdx.x * 16;
  const int nSub  = blockIdx.y * 64 + wave * 16;
  const int mRow  = mBase + (lane & 15);
  const int kA    = (lane >> 4) * 8;    // A: lanes 16-31 carry the K+8 block
  const int nCol  = nSub + (lane & 15);
  const int kB    = (lane >> 4) * 8;    // B: lanes 16-31 carry the K+8 block

  // per-lane base pointers (strength-reduced; bumped by constant stride per step)
  const _Float16* aP16 = (const _Float16*)Av + (size_t)mRow * lda + kA;
  const float*    aP32 = (const float*)Av    + (size_t)mRow * lda + kA;
  const _Float16* bP   = BT + (size_t)nCol * ldb + kB;

  v8f acc0 = {}, acc1 = {};

  #pragma unroll 2
  for (int kc = 0; kc < KTOT; kc += 64) {
    AFrag a0, a1, b0, b1;
    if (A_F16) {
      a0.h[0] = *(const v8h*)(aP16);
      a0.h[1] = *(const v8h*)(aP16 + 16);
      a1.h[0] = *(const v8h*)(aP16 + 32);
      a1.h[1] = *(const v8h*)(aP16 + 48);
      aP16 += 64;
    } else {
      a0.h[0] = __builtin_convertvector(*(const v8f*)(aP32),      v8h);
      a0.h[1] = __builtin_convertvector(*(const v8f*)(aP32 + 16), v8h);
      a1.h[0] = __builtin_convertvector(*(const v8f*)(aP32 + 32), v8h);
      a1.h[1] = __builtin_convertvector(*(const v8f*)(aP32 + 48), v8h);
      aP32 += 64;
    }
    b0.h[0] = *(const v8h*)(bP);
    b0.h[1] = *(const v8h*)(bP + 16);
    b1.h[0] = *(const v8h*)(bP + 32);
    b1.h[1] = *(const v8h*)(bP + 48);
    bP += 64;
    acc0 = __builtin_amdgcn_wmma_f32_16x16x32_f16(false, a0.v, false, b0.v,
                                                  (short)0, acc0, false, false);
    acc1 = __builtin_amdgcn_wmma_f32_16x16x32_f16(false, a1.v, false, b1.v,
                                                  (short)0, acc1, false, false);
  }
  #pragma unroll
  for (int r = 0; r < 8; ++r) acc0[r] += acc1[r];

  if (MODE == 2) {
    const float bcol = bias[nCol];
    #pragma unroll
    for (int r = 0; r < 8; ++r) {
      const int row = mBase + r + (lane >> 4) * 8;
      const float rc = rcnt[row];                       // reciprocal count (0 if none)
      const float vb = (rc != 0.0f) ? bcol : 0.0f;
      C[(size_t)row * ldc + nCol] = fmaf(acc0[r], rc, vb);   // no division
    }
  } else if (MODE == 1) {
    #pragma unroll
    for (int r = 0; r < 8; ++r) {
      const int row = mBase + r + (lane >> 4) * 8;
      C[(size_t)row * ldc + nCol] += acc0[r];
    }
  } else {
    const float bcol = bias[nCol];
    #pragma unroll
    for (int r = 0; r < 8; ++r) {
      const int row = mBase + r + (lane >> 4) * 8;
      C[(size_t)row * ldc + nCol] = acc0[r] + bcol;
    }
  }
}

// ---------------- K4: sigmoid gate + combine + RMSNorm per row ----------------
__global__ __launch_bounds__(256)
void fuse_kernel(const float* __restrict__ x, const float* __restrict__ gf,
                 const float* __restrict__ gb, const float* __restrict__ logits,
                 const float* __restrict__ scale, float* __restrict__ out) {
  __shared__ float hbuf[D_DIM];
  __shared__ float red[256];
  const int m = blockIdx.x;
  const int t = threadIdx.x;
  float lss = 0.f;
  for (int d = t; d < D_DIM; d += 256) {
    const size_t idx = (size_t)m * D_DIM + d;
    const float a  = 1.0f / (1.0f + __expf(-logits[idx]));
    const float gc = 0.5f * (gf[idx] + gb[idx]);
    const float h  = a * x[idx] + (1.0f - a) * gc;
    hbuf[d] = h;
    lss += h * h;
  }
  red[t] = lss; __syncthreads();
  for (int s = 128; s > 0; s >>= 1) { if (t < s) red[t] += red[t + s]; __syncthreads(); }
  const float inv = rsqrtf(red[0] / (float)D_DIM + 1e-5f);
  for (int d = t; d < D_DIM; d += 256)
    out[(size_t)m * D_DIM + d] = hbuf[d] * inv * scale[d];
}

// ---------------- host-side orchestration ----------------
extern "C" void kernel_launch(void* const* d_in, const int* in_sizes, int n_in,
                              void* d_out, int out_size, void* d_ws, size_t ws_size,
                              hipStream_t stream) {
  (void)in_sizes; (void)n_in; (void)out_size; (void)ws_size;
  const float* x    = (const float*)d_in[0];
  const float* W_dr = (const float*)d_in[1];
  const float* b_dr = (const float*)d_in[2];
  const float* Wf   = (const float*)d_in[3];
  const float* bf   = (const float*)d_in[4];
  const float* Wb   = (const float*)d_in[5];
  const float* bb   = (const float*)d_in[6];
  const float* Wg   = (const float*)d_in[7];
  const float* bg   = (const float*)d_in[8];
  const float* sc   = (const float*)d_in[9];
  float* out = (float*)d_out;

  uint8_t* w = (uint8_t*)d_ws;
  size_t off = 0;
  auto take = [&](size_t bytes) -> void* {
    void* p = w + off;
    off += (bytes + 255) & ~(size_t)255;
    return p;
  };
  float*    z      = (float*)   take((size_t)MTOT * R_DIM * 4);
  _Float16* Sf     = (_Float16*)take((size_t)MTOT * KPAD * 2);
  _Float16* Sb     = (_Float16*)take((size_t)MTOT * KPAD * 2);
  float*    rcntf  = (float*)   take((size_t)MTOT * 4);
  float*    rcntb  = (float*)   take((size_t)MTOT * 4);
  _Float16* WfT    = (_Float16*)take((size_t)D_DIM * KPAD * 2);
  _Float16* WbT    = (_Float16*)take((size_t)D_DIM * KPAD * 2);
  _Float16* WgT    = (_Float16*)take((size_t)D_DIM * E_DIM * 2);
  float*    gfwd   = (float*)   take((size_t)MTOT * D_DIM * 4);
  float*    gbwd   = (float*)   take((size_t)MTOT * D_DIM * 4);
  float*    logits = (float*)   take((size_t)MTOT * D_DIM * 4);

  // weight prep: transpose + f16 convert + K-pad
  transpose_w_kernel<<<(D_DIM * KPAD + 255) / 256, 256, 0, stream>>>(Wf, WfT, P_DIM, KPAD, D_DIM);
  transpose_w_kernel<<<(D_DIM * KPAD + 255) / 256, 256, 0, stream>>>(Wb, WbT, P_DIM, KPAD, D_DIM);
  transpose_w_kernel<<<(D_DIM * E_DIM + 255) / 256, 256, 0, stream>>>(Wg, WgT, E_DIM, E_DIM, D_DIM);

  // z projection
  zproj_kernel<<<MTOT / 8, 256, 0, stream>>>(x, W_dr, b_dr, z);

  // summed normalized pluecker vectors per direction
  plucker_kernel<<<MTOT, 256, 0, stream>>>(z, Sf, rcntf, 0);
  plucker_kernel<<<MTOT, 256, 0, stream>>>(z, Sb, rcntb, 1);

  // GEMMs: grid = (M/16 tiles, N/64 tiles), 4 waves per WG
  dim3 gg(MTOT / 16, D_DIM / 64);
  // directional: g = (S @ WpT)*rcnt + bias  (f16 A, K=512)
  gemm_wmma_kernel<1, 2, KPAD><<<gg, 128, 0, stream>>>(Sf, KPAD, WfT, KPAD, bf, rcntf, gfwd, D_DIM);
  gemm_wmma_kernel<1, 2, KPAD><<<gg, 128, 0, stream>>>(Sb, KPAD, WbT, KPAD, bb, rcntb, gbwd, D_DIM);
  // gate logits: x@Wg[0:768] + g_fwd@Wg[768:1536] + g_bwd@Wg[1536:2304] + bg  (f32 A, K=768)
  gemm_wmma_kernel<0, 0, D_DIM><<<gg, 128, 0, stream>>>(x,    D_DIM, WgT,             E_DIM, bg,      nullptr, logits, D_DIM);
  gemm_wmma_kernel<0, 1, D_DIM><<<gg, 128, 0, stream>>>(gfwd, D_DIM, WgT + D_DIM,     E_DIM, nullptr, nullptr, logits, D_DIM);
  gemm_wmma_kernel<0, 1, D_DIM><<<gg, 128, 0, stream>>>(gbwd, D_DIM, WgT + 2 * D_DIM, E_DIM, nullptr, nullptr, logits, D_DIM);

  // gate + combine + rmsnorm
  fuse_kernel<<<MTOT, 256, 0, stream>>>(x, gfwd, gbwd, logits, sc, out);
}
